// SampledMixtureOfSoftmaxes_24429773979775
// MI455X (gfx1250) — matrix operand order repaired and checked by reading, not compiled, not run on hardware
//
#include <hip/hip_runtime.h>
#include <math.h>

typedef __attribute__((ext_vector_type(16))) _Float16 v16h;
typedef __attribute__((ext_vector_type(8)))  _Float16 v8h;
typedef __attribute__((ext_vector_type(8)))  float    v8f;

#define BB   1024
#define EE   64
#define HH   4
#define VV   1000000
#define SS   22222
#define SPAD 22224            // S rounded up to 16
#define SP1  (SS + 1)         // 22223, probs row length
#define MM   (BB * HH)        // 4096 GEMM rows
#define NT   (SPAD / 16)      // 1389 N-tiles
#define NC   12               // S-chunks per M-super-tile
#define TPC  ((NT + NC - 1) / NC)  // 116 tiles per chunk
#define MSUP (MM / 32)        // 128 M-super-tiles (2 x 16 rows each)

__device__ __forceinline__ float device_log_q() {
  // log(-expm1(S * log1p(-1/V))) in double precision
  double p = 1.0 / (double)VV;
  return (float)log(-expm1((double)SS * log1p(-p)));
}

// ---------------- prep: mos_proj (f16 A-matrix), pi, true-logit shift ----------------
__global__ __launch_bounds__(256)
void k_prep_mos(const float* __restrict__ inputs, const float* __restrict__ emb,
                const float* __restrict__ proj,  const float* __restrict__ mixm,
                const float* __restrict__ biases, const int* __restrict__ label,
                _Float16* __restrict__ mosA, float* __restrict__ tshift,
                float* __restrict__ piF) {
  __shared__ float sin[EE];
  __shared__ float smos[HH * EE];
  __shared__ float sml[HH];
  __shared__ float stl[HH];
  const int b = blockIdx.x;
  const int t = threadIdx.x;
  if (t < EE) sin[t] = inputs[b * EE + t];
  __syncthreads();
  float acc = 0.f;
  const float* pr = proj + t * EE;
  #pragma unroll 8
  for (int e = 0; e < EE; ++e) acc += sin[e] * pr[e];
  float m = tanhf(acc);
  smos[t] = m;
  mosA[(b * HH + (t >> 6)) * EE + (t & 63)] = (_Float16)m;
  __syncthreads();
  if (t < HH) {
    const int lab = label[b];
    const float* w  = emb + (long long)lab * EE;
    const float* mr = smos + t * EE;
    float tl = 0.f, ml = 0.f;
    for (int e = 0; e < EE; ++e) { tl += mr[e] * w[e]; ml += sin[e] * mixm[t * EE + e]; }
    stl[t] = tl + biases[lab] - device_log_q();
    sml[t] = ml;
  }
  __syncthreads();
  if (t == 0) {
    float mx = fmaxf(fmaxf(sml[0], sml[1]), fmaxf(sml[2], sml[3]));
    float ex[HH], d = 0.f;
    #pragma unroll
    for (int h = 0; h < HH; ++h) { ex[h] = __expf(sml[h] - mx); d += ex[h]; }
    #pragma unroll
    for (int h = 0; h < HH; ++h) {
      piF[b * HH + h]    = ex[h] / d;
      tshift[b * HH + h] = stl[h];
    }
  }
}

// ---------------- prep: sampled embedding gather -> f16 B-matrix [SPAD][64] ----------------
__global__ __launch_bounds__(256)
void k_prep_samp(const float* __restrict__ emb, const float* __restrict__ biases,
                 const int* __restrict__ sids,
                 _Float16* __restrict__ sampB, float* __restrict__ biasS) {
  const int idx = blockIdx.x * blockDim.x + threadIdx.x;  // SPAD*16 threads
  if (idx >= SPAD * 16) return;
  const int s = idx >> 4, j = idx & 15;
  _Float16* d = sampB + s * EE + 4 * j;
  if (s < SS) {
    const int id = sids[s];
    const float4 w = *(const float4*)(emb + (long long)id * EE + 4 * j);
    d[0] = (_Float16)w.x; d[1] = (_Float16)w.y;
    d[2] = (_Float16)w.z; d[3] = (_Float16)w.w;
    if (j == 0) biasS[s] = biases[id] - device_log_q();
  } else {
    d[0] = d[1] = d[2] = d[3] = (_Float16)0.f;
    if (j == 0) biasS[s] = -1e30f;  // exp -> 0, pad column inert
  }
}

// Build a v16h fragment from two contiguous 8-half runs
__device__ __forceinline__ v16h load16(const _Float16* p0, const _Float16* p1) {
  v8h a = *(const v8h*)p0;
  v8h b = *(const v8h*)p1;
  v16h r;
  #pragma unroll
  for (int i = 0; i < 8; ++i) { r[i] = a[i]; r[i + 8] = b[i]; }
  return r;
}

// Per-ISA 16-bit A 16x32 layout: lanes 0-15 row m hold K{0-7,16-23}, lanes 16-31 hold K{8-15,24-31}
__device__ __forceinline__ v16h loadA(const _Float16* Arow, int hi, int kb) {
  return load16(Arow + kb * 32 + hi * 8, Arow + kb * 32 + 16 + hi * 8);
}
// 16-bit B 32x16 layout: lane n<16 holds col n K{0-7,16-23}, lane n+16 holds col n K{8-15,24-31} (matches A)
__device__ __forceinline__ v16h loadB(const _Float16* Brow, int hi, int kb) {
  return load16(Brow + kb * 32 + hi * 8, Brow + kb * 32 + 16 + hi * 8);
}

// ---------------- pass 1: sum of exp(logit + bias - tshift) per row ----------------
// Each wave owns TWO 16-row M-tiles and reuses every B fragment for both (halves L2 traffic).
__global__ __launch_bounds__(128)
void k_pass1(const _Float16* __restrict__ mosA, const _Float16* __restrict__ sampB,
             const float* __restrict__ biasS, const float* __restrict__ tshift,
             float* __restrict__ part) {
  __shared__ float red[4][32];
  const int mt = blockIdx.x, ch = blockIdx.y;   // mt: 0..MSUP-1
  const int wave = threadIdx.x >> 5, lane = threadIdx.x & 31;
  const int lo = lane & 15, hi = lane >> 4;
  const int R0 = mt * 32, R1 = R0 + 16;
  const _Float16* Arow0 = mosA + (R0 + lo) * EE;
  const _Float16* Arow1 = mosA + (R1 + lo) * EE;
  const v16h a00 = loadA(Arow0, hi, 0);
  const v16h a01 = loadA(Arow0, hi, 1);
  const v16h a10 = loadA(Arow1, hi, 0);
  const v16h a11 = loadA(Arow1, hi, 1);
  float tsh[16], sum[16];
  #pragma unroll
  for (int j = 0; j < 8; ++j) {
    tsh[j]     = tshift[R0 + 8 * hi + j];
    tsh[j + 8] = tshift[R1 + 8 * hi + j];
    sum[j] = 0.f; sum[j + 8] = 0.f;
  }
  const int t0 = ch * TPC, t1 = min(t0 + TPC, NT);
  for (int nt = t0 + wave; nt < t1; nt += 4) {
    const int s = nt * 16 + lo;
    const _Float16* Brow = sampB + s * EE;
    const v16h b0 = loadB(Brow, hi, 0);
    const v16h b1 = loadB(Brow, hi, 1);
    const float bS = biasS[s];
    v8f acc0 = {}, acc1 = {};
    // interleave independent chains so dependent WMMAs are separated
    acc0 = __builtin_amdgcn_wmma_f32_16x16x32_f16(false, a00, false, b0, (short)0, acc0, false, false);
    acc1 = __builtin_amdgcn_wmma_f32_16x16x32_f16(false, a10, false, b0, (short)0, acc1, false, false);
    acc0 = __builtin_amdgcn_wmma_f32_16x16x32_f16(false, a01, false, b1, (short)0, acc0, false, false);
    acc1 = __builtin_amdgcn_wmma_f32_16x16x32_f16(false, a11, false, b1, (short)0, acc1, false, false);
    #pragma unroll
    for (int j = 0; j < 8; ++j) {
      sum[j]     += __expf(acc0[j] + bS - tsh[j]);
      sum[j + 8] += __expf(acc1[j] + bS - tsh[j + 8]);
    }
  }
  // reduce columns across each 16-lane group (xor masks stay within group)
  #pragma unroll
  for (int j = 0; j < 16; ++j) {
    float v = sum[j];
    v += __shfl_xor(v, 1, 32);
    v += __shfl_xor(v, 2, 32);
    v += __shfl_xor(v, 4, 32);
    v += __shfl_xor(v, 8, 32);
    sum[j] = v;
  }
  if (lo == 0) {
    #pragma unroll
    for (int j = 0; j < 8; ++j) {
      red[wave][8 * hi + j]      = sum[j];       // rows R0+8hi+j
      red[wave][16 + 8 * hi + j] = sum[j + 8];   // rows R1+8hi+j
    }
  }
  __syncthreads();
  if (threadIdx.x < 32) {
    const float tot = red[0][threadIdx.x] + red[1][threadIdx.x] +
                      red[2][threadIdx.x] + red[3][threadIdx.x];
    part[(R0 + threadIdx.x) * NC + ch] = tot;    // rows R0..R0+31
  }
}

// ---------------- combine partials: coef = pi / gsum  (gsum includes exp(true - t) = 1) ----------------
__global__ __launch_bounds__(256)
void k_combine(const float* __restrict__ part, const float* __restrict__ piF,
               float* __restrict__ coef) {
  const int r = blockIdx.x * blockDim.x + threadIdx.x;
  if (r >= MM) return;
  float g = 1.f;
  #pragma unroll
  for (int c = 0; c < NC; ++c) g += part[r * NC + c];
  coef[r] = piF[r] / g;
}

// ---------------- probs[:,0] and loss ----------------
__global__ __launch_bounds__(256)
void k_head(const float* __restrict__ coef, float* __restrict__ out) {
  __shared__ float red[256];
  const int t = threadIdx.x;
  float lacc = 0.f;
  for (int b = t; b < BB; b += 256) {
    const float p0 = coef[4 * b] + coef[4 * b + 1] + coef[4 * b + 2] + coef[4 * b + 3];
    out[(long long)b * SP1] = p0;
    lacc -= logf(p0);
  }
  red[t] = lacc;
  __syncthreads();
  for (int s = 128; s > 0; s >>= 1) { if (t < s) red[t] += red[t + s]; __syncthreads(); }
  if (t == 0) out[(long long)BB * SP1] = red[0] / (float)BB;
}

// ---------------- pass 2: recompute logits, write mixture probs (2 M-tiles per wave) ----------------
__global__ __launch_bounds__(128)
void k_pass2(const _Float16* __restrict__ mosA, const _Float16* __restrict__ sampB,
             const float* __restrict__ biasS, const float* __restrict__ tshift,
             const float* __restrict__ coef, float* __restrict__ out) {
  const int mt = blockIdx.x, ch = blockIdx.y;
  const int wave = threadIdx.x >> 5, lane = threadIdx.x & 31;
  const int lo = lane & 15, hi = lane >> 4;
  const int R0 = mt * 32, R1 = R0 + 16;
  const _Float16* Arow0 = mosA + (R0 + lo) * EE;
  const _Float16* Arow1 = mosA + (R1 + lo) * EE;
  const v16h a00 = loadA(Arow0, hi, 0);
  const v16h a01 = loadA(Arow0, hi, 1);
  const v16h a10 = loadA(Arow1, hi, 0);
  const v16h a11 = loadA(Arow1, hi, 1);
  float tsh[16], cf[16];
  #pragma unroll
  for (int j = 0; j < 8; ++j) {
    tsh[j]     = tshift[R0 + 8 * hi + j];
    tsh[j + 8] = tshift[R1 + 8 * hi + j];
    cf[j]      = coef[R0 + 8 * hi + j];
    cf[j + 8]  = coef[R1 + 8 * hi + j];
  }
  const int bb0 = (R0 + 8 * hi) >> 2;  // tile0: rows j=0..3 -> bb0, j=4..7 -> bb0+1
  const int bb2 = (R1 + 8 * hi) >> 2;  // tile1
  const int t0 = ch * TPC, t1 = min(t0 + TPC, NT);
  for (int nt = t0 + wave; nt < t1; nt += 4) {
    const int s = nt * 16 + lo;
    const _Float16* Brow = sampB + s * EE;
    const v16h b0 = loadB(Brow, hi, 0);
    const v16h b1 = loadB(Brow, hi, 1);
    const float bS = biasS[s];
    v8f acc0 = {}, acc1 = {};
    acc0 = __builtin_amdgcn_wmma_f32_16x16x32_f16(false, a00, false, b0, (short)0, acc0, false, false);
    acc1 = __builtin_amdgcn_wmma_f32_16x16x32_f16(false, a10, false, b0, (short)0, acc1, false, false);
    acc0 = __builtin_amdgcn_wmma_f32_16x16x32_f16(false, a01, false, b1, (short)0, acc0, false, false);
    acc1 = __builtin_amdgcn_wmma_f32_16x16x32_f16(false, a11, false, b1, (short)0, acc1, false, false);
    if (s < SS) {
      const float v0 = __expf(acc0[0] + bS - tsh[0]) * cf[0] + __expf(acc0[1] + bS - tsh[1]) * cf[1]
                     + __expf(acc0[2] + bS - tsh[2]) * cf[2] + __expf(acc0[3] + bS - tsh[3]) * cf[3];
      const float v1 = __expf(acc0[4] + bS - tsh[4]) * cf[4] + __expf(acc0[5] + bS - tsh[5]) * cf[5]
                     + __expf(acc0[6] + bS - tsh[6]) * cf[6] + __expf(acc0[7] + bS - tsh[7]) * cf[7];
      const float v2 = __expf(acc1[0] + bS - tsh[8])  * cf[8]  + __expf(acc1[1] + bS - tsh[9])  * cf[9]
                     + __expf(acc1[2] + bS - tsh[10]) * cf[10] + __expf(acc1[3] + bS - tsh[11]) * cf[11];
      const float v3 = __expf(acc1[4] + bS - tsh[12]) * cf[12] + __expf(acc1[5] + bS - tsh[13]) * cf[13]
                     + __expf(acc1[6] + bS - tsh[14]) * cf[14] + __expf(acc1[7] + bS - tsh[15]) * cf[15];
      out[(long long)bb0 * SP1 + 1 + s]       = v0;
      out[(long long)(bb0 + 1) * SP1 + 1 + s] = v1;
      out[(long long)bb2 * SP1 + 1 + s]       = v2;
      out[(long long)(bb2 + 1) * SP1 + 1 + s] = v3;
    }
  }
}

extern "C" void kernel_launch(void* const* d_in, const int* in_sizes, int n_in,
                              void* d_out, int out_size, void* d_ws, size_t ws_size,
                              hipStream_t stream) {
  const float* inputs = (const float*)d_in[0];
  const float* emb    = (const float*)d_in[1];
  const float* proj   = (const float*)d_in[2];
  const float* mixm   = (const float*)d_in[3];
  const float* biases = (const float*)d_in[4];
  const int*   label  = (const int*)d_in[5];
  const int*   sids   = (const int*)d_in[6];
  float* out = (float*)d_out;

  // workspace carve-up (all 16B-aligned)
  _Float16* mosA  = (_Float16*)d_ws;                 // MM*EE halves   = 512 KB
  _Float16* sampB = mosA + (size_t)MM * EE;          // SPAD*EE halves = 2.78 MB
  float* biasS    = (float*)(sampB + (size_t)SPAD * EE);
  float* tshift   = biasS + SPAD;
  float* piF      = tshift + MM;
  float* coef     = piF + MM;
  float* part     = coef + MM;                       // MM*NC floats = 192 KB

  k_prep_mos<<<BB, 256, 0, stream>>>(inputs, emb, proj, mixm, biases, label,
                                     mosA, tshift, piF);
  k_prep_samp<<<(SPAD * 16) / 256, 256, 0, stream>>>(emb, biases, sids, sampB, biasS);
  k_pass1<<<dim3(MSUP, NC), 128, 0, stream>>>(mosA, sampB, biasS, tshift, part);
  k_combine<<<MM / 256, 256, 0, stream>>>(part, piF, coef);
  k_head<<<1, 256, 0, stream>>>(coef, out);
  k_pass2<<<dim3(MSUP, NC), 128, 0, stream>>>(mosA, sampB, biasS, tshift, coef, out);
}
